// MyHierAttNet_37821482008870
// MI455X (gfx1250) — matrix operand order repaired and checked by reading, not compiled, not run on hardware
//
#include <hip/hip_runtime.h>
#include <math.h>

// ---------------------------------------------------------------------------
// MI455X / gfx1250 implementation of the hierarchical attention network.
// All GEMM work (BiGRU input+recurrent projections, attention projections)
// runs on v_wmma_f32_16x16x32_bf16 with f32 accumulation.
// ---------------------------------------------------------------------------

typedef __attribute__((ext_vector_type(16))) __bf16 v16bf;
typedef __attribute__((ext_vector_type(8)))  float  v8f;

namespace {
constexpr int Ev = 300, Hv = 150, Bv = 64, Sv = 16, Tv = 48, NCv = 300;
constexpr int HP = 160;      // hidden padded 150 -> 160 (10 x 16)
constexpr int EP = 320;      // input  padded 300 -> 320 (10 x 32)
constexpr int G2 = 300;      // 2H
constexpr int GP = 320;      // 2H padded
constexpr int N1 = Bv * Sv;  // 1024  word-level batch
constexpr int N2 = Bv;       // 64    sentence-level batch
constexpr int WPB = 8, TPB = WPB * 32;   // 8 wave32s per block
}

// ---------------- WMMA fragment helpers (CDNA5 bf16 16x16x32 layouts) -------

__device__ __forceinline__ float sigf(float x) { return 1.0f / (1.0f + expf(-x)); }

// A fragment: 16x32 bf16, A row-major (M x K). lane 0-15: M=lane, K=k0..k0+15;
// lane 16-31: M=lane-16, K=k0+16..k0+31. element i -> K=16*(lane/16)+i.
__device__ __forceinline__ v16bf ldA(const __bf16* A, int lda, int row0, int k0) {
  int lane = threadIdx.x & 31;
  const __bf16* p = A + (size_t)(row0 + (lane & 15)) * lda + k0 + (lane >> 4) * 16;
  v16bf a;
#pragma unroll
  for (int i = 0; i < 8; ++i) a[i] = p[i];
#pragma unroll
  for (int i = 0; i < 8; ++i) a[8 + i] = p[8 + i];
  return a;
}

// B fragment: 32x16 bf16 where B = W^T and W is row-major (N x K).
// lane 0-15: N=lane, K=k0..k0+15 ; lane 16-31: N=lane-16, K=k0+16..k0+31.
__device__ __forceinline__ v16bf ldB(const __bf16* W, int ldw, int col0, int k0) {
  int lane = threadIdx.x & 31;
  const __bf16* p = W + (size_t)(col0 + (lane & 15)) * ldw + k0 + (lane >> 4) * 16;
  v16bf b;
#pragma unroll
  for (int i = 0; i < 16; ++i) b[i] = p[i];
  return b;
}

__device__ __forceinline__ v8f wmma_bf16(v16bf a, v16bf b, v8f c) {
  return __builtin_amdgcn_wmma_f32_16x16x32_bf16(false, a, false, b, (short)0, c,
                                                 false, false);
}

// ---------------- packing kernels (f32 -> zero-padded bf16 / f32) -----------

__global__ void k_pack_bf16(const float* __restrict__ src, __bf16* __restrict__ dst,
                            int nmat, int sr, int sc, int dr, int dc) {
  size_t tot = (size_t)nmat * dr * dc;
  for (size_t i = (size_t)blockIdx.x * blockDim.x + threadIdx.x; i < tot;
       i += (size_t)gridDim.x * blockDim.x) {
    int c = (int)(i % dc);
    size_t q = i / dc;
    int r = (int)(q % dr);
    int m = (int)(q / dr);
    float v = (r < sr && c < sc) ? src[((size_t)m * sr + r) * sc + c] : 0.0f;
    dst[i] = (__bf16)v;
  }
}

__global__ void k_pack_f32(const float* __restrict__ src, float* __restrict__ dst,
                           int nmat, int sr, int dr) {
  size_t tot = (size_t)nmat * dr;
  for (size_t i = (size_t)blockIdx.x * blockDim.x + threadIdx.x; i < tot;
       i += (size_t)gridDim.x * blockDim.x) {
    int r = (int)(i % dr);
    int m = (int)(i / dr);
    dst[i] = (r < sr) ? src[(size_t)m * sr + r] : 0.0f;
  }
}

// embedding gather + transpose to (T, N1, EP) bf16, zero padded
__global__ void k_pack_words(const int* __restrict__ docs,
                             const float* __restrict__ wemb,
                             __bf16* __restrict__ Xw) {
  size_t tot = (size_t)Tv * N1 * EP;
  for (size_t i = (size_t)blockIdx.x * blockDim.x + threadIdx.x; i < tot;
       i += (size_t)gridDim.x * blockDim.x) {
    int e = (int)(i % EP);
    size_t q = i / EP;
    int n = (int)(q % N1);
    int t = (int)(q / N1);
    float v = 0.0f;
    if (e < Ev) {
      int b = n / Sv, s = n % Sv;
      int tok = docs[((size_t)b * Sv + s) * Tv + t];
      v = wemb[(size_t)tok * Ev + e];
    }
    Xw[i] = (__bf16)v;
  }
}

// sent_vec (B,S,300) f32 -> xs (S,B,320) bf16, zero padded (transpose 1,0,2)
__global__ void k_pack_xs(const float* __restrict__ sv, __bf16* __restrict__ xsb) {
  size_t tot = (size_t)Sv * N2 * GP;
  for (size_t i = (size_t)blockIdx.x * blockDim.x + threadIdx.x; i < tot;
       i += (size_t)gridDim.x * blockDim.x) {
    int g = (int)(i % GP);
    size_t q = i / GP;
    int b = (int)(q % N2);
    int s = (int)(q / N2);
    float v = (g < G2) ? sv[((size_t)b * Sv + s) * G2 + g] : 0.0f;
    xsb[i] = (__bf16)v;
  }
}

// ---------------- fused GRU step (both projections + gate math) -------------
// One wave computes one 16x16 tile of h_new.  4 WMMA accumulator chains:
//  acc_r = gi_r + gh_r, acc_z = gi_z + gh_z, acc_in = gi_n, acc_hn = gh_n.
__global__ void k_gru_step(const __bf16* __restrict__ Xt, int Kxp,
                           const __bf16* __restrict__ Hprev,
                           const __bf16* __restrict__ Wx,  // (3, HP, Kxp)
                           const __bf16* __restrict__ Wh,  // (3, HP, HP)
                           const float* __restrict__ bih,  // (3*HP)
                           const float* __restrict__ bhh,  // (3*HP)
                           int N,
                           __bf16* __restrict__ Hnew,            // (N, HP)
                           float* __restrict__ hsOut, int hsStride,
                           __bf16* __restrict__ hsbOut, int hsbStride) {
  int wave = (int)((blockIdx.x * blockDim.x + threadIdx.x) >> 5);
  const int ctn = HP / 16;  // 10 column tiles
  int nT = (N / 16) * ctn;
  if (wave >= nT) return;
  int row0 = (wave / ctn) * 16;
  int col0 = (wave % ctn) * 16;

  const __bf16* Wxr = Wx;
  const __bf16* Wxz = Wx + (size_t)HP * Kxp;
  const __bf16* Wxn = Wx + (size_t)2 * HP * Kxp;
  const __bf16* Whr = Wh;
  const __bf16* Whz = Wh + (size_t)HP * HP;
  const __bf16* Whn = Wh + (size_t)2 * HP * HP;

  v8f accr = {}, accz = {}, accin = {}, acchn = {};

  for (int k = 0; k < Kxp; k += 32) {
    v16bf a = ldA(Xt, Kxp, row0, k);
    __builtin_prefetch(Xt + (size_t)row0 * Kxp + k + 64, 0, 1);
    accr  = wmma_bf16(a, ldB(Wxr, Kxp, col0, k), accr);
    accz  = wmma_bf16(a, ldB(Wxz, Kxp, col0, k), accz);
    accin = wmma_bf16(a, ldB(Wxn, Kxp, col0, k), accin);
  }
  for (int k = 0; k < HP; k += 32) {
    v16bf a = ldA(Hprev, HP, row0, k);
    accr  = wmma_bf16(a, ldB(Whr, HP, col0, k), accr);
    accz  = wmma_bf16(a, ldB(Whz, HP, col0, k), accz);
    acchn = wmma_bf16(a, ldB(Whn, HP, col0, k), acchn);
  }

  int lane = threadIdx.x & 31;
  int kh = lane >> 4;
  int j = col0 + (lane & 15);
  float br = bih[j] + bhh[j];
  float bz = bih[HP + j] + bhh[HP + j];
  float bin = bih[2 * HP + j];
  float bhn = bhh[2 * HP + j];
#pragma unroll
  for (int r = 0; r < 8; ++r) {
    int n = row0 + r + 8 * kh;
    float hp = (float)Hprev[(size_t)n * HP + j];
    float rv = sigf(accr[r] + br);
    float zv = sigf(accz[r] + bz);
    float nv = tanhf(accin[r] + bin + rv * (acchn[r] + bhn));
    float h2 = (1.0f - zv) * nv + zv * hp;
    Hnew[(size_t)n * HP + j] = (__bf16)h2;
    if (j < Hv) {
      hsOut[(size_t)n * hsStride + j] = h2;
      hsbOut[(size_t)n * hsbStride + j] = (__bf16)h2;
    }
  }
}

// ---------------- attention projection: U = tanh(A @ W^T + b), bf16 out -----
__global__ void k_gemm_tanh(const __bf16* __restrict__ A,     // (M, Kp)
                            const __bf16* __restrict__ W,     // (GP, Kp)
                            const float* __restrict__ bias,   // (GP)
                            int M, int Kp, __bf16* __restrict__ U) {
  int wave = (int)((blockIdx.x * blockDim.x + threadIdx.x) >> 5);
  const int ctn = GP / 16;  // 20
  int nT = (M / 16) * ctn;
  if (wave >= nT) return;
  int row0 = (wave / ctn) * 16;
  int col0 = (wave % ctn) * 16;
  v8f acc = {};
  for (int k = 0; k < Kp; k += 32) {
    v16bf a = ldA(A, Kp, row0, k);
    v16bf b = ldB(W, Kp, col0, k);
    acc = wmma_bf16(a, b, acc);
  }
  int lane = threadIdx.x & 31;
  int kh = lane >> 4;
  int j = col0 + (lane & 15);
  float bj = bias[j];
#pragma unroll
  for (int r = 0; r < 8; ++r) {
    int m = row0 + r + 8 * kh;
    U[(size_t)m * GP + j] = (__bf16)tanhf(acc[r] + bj);
  }
}

// scores[m] = sum_j U[m,j] * ctx[j]   (ctx zero-padded to GP)
__global__ void k_att_scores(const __bf16* __restrict__ U,
                             const float* __restrict__ ctx, int M,
                             float* __restrict__ scores) {
  int row = (int)((blockIdx.x * blockDim.x + threadIdx.x) >> 5);
  if (row >= M) return;
  int lane = threadIdx.x & 31;
  float s = 0.0f;
  for (int j = lane; j < GP; j += 32) s += (float)U[(size_t)row * GP + j] * ctx[j];
#pragma unroll
  for (int o = 16; o > 0; o >>= 1) s += __shfl_xor(s, o, 32);
  if (lane == 0) scores[row] = s;
}

// softmax over t (axis 0) for sc[t*N + n], in place
__global__ void k_softmax_t(float* __restrict__ sc, int T, int N) {
  int n = blockIdx.x * blockDim.x + threadIdx.x;
  if (n >= N) return;
  float m = -3.4e38f;
  for (int t = 0; t < T; ++t) m = fmaxf(m, sc[(size_t)t * N + n]);
  float sum = 0.0f;
  for (int t = 0; t < T; ++t) {
    float e = expf(sc[(size_t)t * N + n] - m);
    sc[(size_t)t * N + n] = e;
    sum += e;
  }
  float inv = 1.0f / sum;
  for (int t = 0; t < T; ++t) sc[(size_t)t * N + n] *= inv;
}

// out[n,g] = sum_t a[t*N+n] * hs[(t*N+n)*G2 + g]
__global__ void k_wsum(const float* __restrict__ a, const float* __restrict__ hs,
                       int T, int N, float* __restrict__ out) {
  int i = blockIdx.x * blockDim.x + threadIdx.x;
  if (i >= N * G2) return;
  int n = i / G2, g = i % G2;
  float s = 0.0f;
  for (int t = 0; t < T; ++t)
    s += a[(size_t)t * N + n] * hs[((size_t)t * N + n) * G2 + g];
  out[i] = s;
}

__global__ void k_gather_sen(const float* __restrict__ sv, const int* __restrict__ sens,
                             float* __restrict__ out) {
  int i = blockIdx.x * blockDim.x + threadIdx.x;
  if (i >= Bv * G2) return;
  int b = i / G2, g = i % G2;
  out[i] = sv[((size_t)b * Sv + sens[b]) * G2 + g];
}

__global__ void k_norms(const float* __restrict__ act, float* __restrict__ out,
                        int nrows) {
  int w = (int)((blockIdx.x * blockDim.x + threadIdx.x) >> 5);
  if (w >= nrows) return;
  int lane = threadIdx.x & 31;
  float s = 0.0f;
  for (int d = lane; d < G2; d += 32) {
    float v = act[(size_t)w * G2 + d];
    s += v * v;
  }
#pragma unroll
  for (int o = 16; o > 0; o >>= 1) s += __shfl_xor(s, o, 32);
  if (lane == 0) out[w] = sqrtf(s);
}

__global__ void k_cossim(const float* __restrict__ act, const float* __restrict__ norms,
                         const int* __restrict__ cand, const float* __restrict__ ent,
                         float* __restrict__ sim, int off) {
  int w = (int)((blockIdx.x * blockDim.x + threadIdx.x) >> 5);
  if (w >= Bv * NCv) return;
  int b = w / NCv, k = w % NCv;
  int lane = threadIdx.x & 31;
  const float* c = ent + (size_t)cand[(size_t)b * NCv + k] * Ev;
  const float* a = act + (size_t)b * G2;
  float num = 0.0f, cn = 0.0f;
  for (int d = lane; d < Ev; d += 32) {
    float cv = c[d];
    num += a[d] * cv;
    cn += cv * cv;
  }
#pragma unroll
  for (int o = 16; o > 0; o >>= 1) {
    num += __shfl_xor(num, o, 32);
    cn += __shfl_xor(cn, o, 32);
  }
  if (lane == 0)
    sim[(size_t)b * 2 * NCv + off + k] = num / fmaxf(norms[b] * sqrtf(cn), 1e-8f);
}

__global__ void k_score(const float* __restrict__ sim, const float* __restrict__ linW,
                        const float* __restrict__ linb, float* __restrict__ score) {
  int i = blockIdx.x * blockDim.x + threadIdx.x;
  if (i >= Bv * NCv) return;
  int b = i / NCv, c = i % NCv;
  const float* sb = sim + (size_t)b * 2 * NCv;
  const float* wc = linW + (size_t)c * 2 * NCv;
  float s = linb[c];
  for (int k = 0; k < 2 * NCv; ++k) s += sb[k] * wc[k];
  score[i] = s;
}

__global__ void k_final(const float* __restrict__ score, const int* __restrict__ idx,
                        float* __restrict__ out) {
  int b = threadIdx.x;
  if (b >= Bv) return;
  const float* sr = score + (size_t)b * NCv;
  out[b] = sr[idx[b] - 1];
  int am = 0;
  float mv = sr[0];
  for (int c = 1; c < NCv; ++c) {
    float v = sr[c];
    if (v > mv) { mv = v; am = c; }
  }
  out[Bv + Bv * NCv + b] = (float)am;
}

// ---------------------------------------------------------------------------

extern "C" void kernel_launch(void* const* d_in, const int* in_sizes, int n_in,
                              void* d_out, int out_size, void* d_ws, size_t ws_size,
                              hipStream_t stream) {
  (void)in_sizes; (void)n_in; (void)out_size; (void)ws_size;

  const int*   docs    = (const int*)d_in[0];
  const int*   sens    = (const int*)d_in[1];
  const int*   cand    = (const int*)d_in[2];
  const int*   idx     = (const int*)d_in[3];
  const float* wemb    = (const float*)d_in[4];
  const float* eemb    = (const float*)d_in[5];
  const float* Wih_w   = (const float*)d_in[6];
  const float* Whh_w   = (const float*)d_in[7];
  const float* bih_w   = (const float*)d_in[8];
  const float* bhh_w   = (const float*)d_in[9];
  const float* attW_w  = (const float*)d_in[10];
  const float* attb_w  = (const float*)d_in[11];
  const float* ctx_w   = (const float*)d_in[12];
  const float* Wih_s   = (const float*)d_in[13];
  const float* Whh_s   = (const float*)d_in[14];
  const float* bih_s   = (const float*)d_in[15];
  const float* bhh_s   = (const float*)d_in[16];
  const float* attW_s  = (const float*)d_in[17];
  const float* attb_s  = (const float*)d_in[18];
  const float* ctx_s   = (const float*)d_in[19];
  const float* linW    = (const float*)d_in[20];
  const float* linb    = (const float*)d_in[21];
  float* out = (float*)d_out;

  // ---- workspace bump allocator (all intermediates stay L2-resident) ----
  size_t off = 0;
  auto alloc = [&](size_t bytes) -> char* {
    char* p = (char*)d_ws + off;
    off = (off + bytes + 255) & ~(size_t)255;
    return p;
  };
  __bf16* Xw     = (__bf16*)alloc((size_t)Tv * N1 * EP * 2);
  __bf16* WihwP  = (__bf16*)alloc((size_t)6 * HP * EP * 2);
  __bf16* WhhwP  = (__bf16*)alloc((size_t)6 * HP * HP * 2);
  __bf16* WihsP  = (__bf16*)alloc((size_t)6 * HP * EP * 2);
  __bf16* WhhsP  = (__bf16*)alloc((size_t)6 * HP * HP * 2);
  float*  bihwP  = (float*)alloc((size_t)6 * HP * 4);
  float*  bhhwP  = (float*)alloc((size_t)6 * HP * 4);
  float*  bihsP  = (float*)alloc((size_t)6 * HP * 4);
  float*  bhhsP  = (float*)alloc((size_t)6 * HP * 4);
  __bf16* attWwP = (__bf16*)alloc((size_t)GP * GP * 2);
  __bf16* attWsP = (__bf16*)alloc((size_t)GP * GP * 2);
  float*  attbwP = (float*)alloc((size_t)GP * 4);
  float*  attbsP = (float*)alloc((size_t)GP * 4);
  float*  ctxwP  = (float*)alloc((size_t)GP * 4);
  float*  ctxsP  = (float*)alloc((size_t)GP * 4);
  __bf16* h0     = (__bf16*)alloc((size_t)N1 * HP * 2);
  __bf16* h1     = (__bf16*)alloc((size_t)N1 * HP * 2);
  float*  hs     = (float*)alloc((size_t)Tv * N1 * G2 * 4);
  __bf16* hsb    = (__bf16*)alloc((size_t)Tv * N1 * GP * 2);
  __bf16* Uw     = (__bf16*)alloc((size_t)Tv * N1 * GP * 2);
  float*  scw    = (float*)alloc((size_t)Tv * N1 * 4);
  float*  svec   = (float*)alloc((size_t)N1 * G2 * 4);
  __bf16* xsb    = (__bf16*)alloc((size_t)Sv * N2 * GP * 2);
  float*  hss    = (float*)alloc((size_t)Sv * N2 * G2 * 4);
  __bf16* hssb   = (__bf16*)alloc((size_t)Sv * N2 * GP * 2);
  __bf16* Us     = (__bf16*)alloc((size_t)Sv * N2 * GP * 2);
  float*  scs    = (float*)alloc((size_t)Sv * N2 * 4);
  float*  osen   = (float*)alloc((size_t)Bv * G2 * 4);
  float*  odoc   = (float*)alloc((size_t)Bv * G2 * 4);
  float*  norms  = (float*)alloc((size_t)2 * Bv * 4);
  float*  sim    = (float*)alloc((size_t)Bv * 2 * NCv * 4);

  // ---- pack inputs: pad + f32->bf16 (gates are contiguous 150-row chunks) --
  k_pack_words<<<4096, 256, 0, stream>>>(docs, wemb, Xw);
  k_pack_bf16<<<1024, 256, 0, stream>>>(Wih_w, WihwP, 6, Hv, Ev, HP, EP);
  k_pack_bf16<<<512, 256, 0, stream>>>(Whh_w, WhhwP, 6, Hv, Hv, HP, HP);
  k_pack_bf16<<<1024, 256, 0, stream>>>(Wih_s, WihsP, 6, Hv, G2, HP, EP);
  k_pack_bf16<<<512, 256, 0, stream>>>(Whh_s, WhhsP, 6, Hv, Hv, HP, HP);
  k_pack_f32<<<4, 256, 0, stream>>>(bih_w, bihwP, 6, Hv, HP);
  k_pack_f32<<<4, 256, 0, stream>>>(bhh_w, bhhwP, 6, Hv, HP);
  k_pack_f32<<<4, 256, 0, stream>>>(bih_s, bihsP, 6, Hv, HP);
  k_pack_f32<<<4, 256, 0, stream>>>(bhh_s, bhhsP, 6, Hv, HP);
  k_pack_bf16<<<512, 256, 0, stream>>>(attW_w, attWwP, 1, G2, G2, GP, GP);
  k_pack_bf16<<<512, 256, 0, stream>>>(attW_s, attWsP, 1, G2, G2, GP, GP);
  k_pack_f32<<<2, 256, 0, stream>>>(attb_w, attbwP, 1, G2, GP);
  k_pack_f32<<<2, 256, 0, stream>>>(attb_s, attbsP, 1, G2, GP);
  k_pack_f32<<<2, 256, 0, stream>>>(ctx_w, ctxwP, 1, G2, GP);
  k_pack_f32<<<2, 256, 0, stream>>>(ctx_s, ctxsP, 1, G2, GP);

  auto run_bigru = [&](const __bf16* X, int Kxp, int N, int T,
                       const __bf16* Wx, const __bf16* Wh,
                       const float* bi, const float* bh,
                       float* hsO, __bf16* hsbO) {
    int tiles = (N / 16) * (HP / 16);
    int blocks = (tiles + WPB - 1) / WPB;
    for (int d = 0; d < 2; ++d) {
      hipMemsetAsync(h0, 0, (size_t)N * HP * sizeof(__bf16), stream);
      __bf16* hp = h0;
      __bf16* hn = h1;
      for (int i = 0; i < T; ++i) {
        int t = (d == 0) ? i : (T - 1 - i);
        k_gru_step<<<blocks, TPB, 0, stream>>>(
            X + (size_t)t * N * Kxp, Kxp, hp,
            Wx + (size_t)d * 3 * HP * Kxp, Wh + (size_t)d * 3 * HP * HP,
            bi + (size_t)d * 3 * HP, bh + (size_t)d * 3 * HP, N, hn,
            hsO + (size_t)t * N * G2 + d * Hv, G2,
            hsbO + (size_t)t * N * GP + d * Hv, GP);
        __bf16* tmp = hp; hp = hn; hn = tmp;
      }
    }
  };

  // ---- word-level BiGRU over T=48, batch 1024 -----------------------------
  hipMemsetAsync(hsb, 0, (size_t)Tv * N1 * GP * sizeof(__bf16), stream);
  run_bigru(Xw, EP, N1, Tv, WihwP, WhhwP, bihwP, bhhwP, hs, hsb);

  // ---- word attention -----------------------------------------------------
  {
    int M = Tv * N1;  // 49152
    int tiles = (M / 16) * (GP / 16);
    k_gemm_tanh<<<(tiles + WPB - 1) / WPB, TPB, 0, stream>>>(hsb, attWwP, attbwP, M, GP, Uw);
    k_att_scores<<<(M + WPB - 1) / WPB, TPB, 0, stream>>>(Uw, ctxwP, M, scw);
    k_softmax_t<<<(N1 + 255) / 256, 256, 0, stream>>>(scw, Tv, N1);
    k_wsum<<<(N1 * G2 + 255) / 256, 256, 0, stream>>>(scw, hs, Tv, N1, svec);
  }
  k_gather_sen<<<(Bv * G2 + 255) / 256, 256, 0, stream>>>(svec, sens, osen);
  k_pack_xs<<<512, 256, 0, stream>>>(svec, xsb);

  // ---- sentence-level BiGRU over T=16, batch 64 ---------------------------
  hipMemsetAsync(hssb, 0, (size_t)Sv * N2 * GP * sizeof(__bf16), stream);
  run_bigru(xsb, GP, N2, Sv, WihsP, WhhsP, bihsP, bhhsP, hss, hssb);

  // ---- doc attention ------------------------------------------------------
  {
    int M = Sv * N2;  // 1024
    int tiles = (M / 16) * (GP / 16);
    k_gemm_tanh<<<(tiles + WPB - 1) / WPB, TPB, 0, stream>>>(hssb, attWsP, attbsP, M, GP, Us);
    k_att_scores<<<(M + WPB - 1) / WPB, TPB, 0, stream>>>(Us, ctxsP, M, scs);
    k_softmax_t<<<1, 256, 0, stream>>>(scs, Sv, N2);
    k_wsum<<<(N2 * G2 + 255) / 256, 256, 0, stream>>>(scs, hss, Sv, N2, odoc);
  }

  // ---- cosine similarity + final linear + gather/argmax -------------------
  k_norms<<<(Bv + WPB - 1) / WPB, TPB, 0, stream>>>(osen, norms, Bv);
  k_norms<<<(Bv + WPB - 1) / WPB, TPB, 0, stream>>>(odoc, norms + Bv, Bv);
  {
    int waves = Bv * NCv;  // 19200
    k_cossim<<<(waves + WPB - 1) / WPB, TPB, 0, stream>>>(osen, norms, cand, eemb, sim, 0);
    k_cossim<<<(waves + WPB - 1) / WPB, TPB, 0, stream>>>(odoc, norms + Bv, cand, eemb, sim, NCv);
  }
  float* score = out + Bv;  // d_out layout: [gold(64)][score(64*300)][argmax(64)]
  k_score<<<(Bv * NCv + 255) / 256, 256, 0, stream>>>(sim, linW, linb, score);
  k_final<<<1, 64, 0, stream>>>(score, idx, out);
}
